// PathConditionalCDEGenerator_28664611733736
// MI455X (gfx1250) — matrix operand order repaired and checked by reading, not compiled, not run on hardware
//
#include <hip/hip_runtime.h>
#include <hip/hip_bf16.h>
#include <stdint.h>

// ============================================================================
// Path-conditional CDE->SDE generator, MI455X (gfx1250) persistent-WGP kernel.
//
//  * kernel 1: convert/pad all MLP weights fp32 -> bf16 into d_ws
//  * kernel 2: ONE persistent block (256 thr = 8 wave32) runs both serial
//    scans with all state in LDS (~307KB of 320KB), every GEMM via
//    v_wmma_f32_16x16x32_bf16.  CDE field weights + per-step dW noise are
//    DMA'd into LDS with the Tensor Data Mover (tensor_load_to_lds +
//    s_wait_tensorcnt); diffusion noise-einsum fused into the big
//    (128x128x2048) layer's tile epilogue for dw_j AND dw_{j+1}.
//    lipswish/tanh use v_rcp_f32 / v_tanh_f32 instead of IEEE div / OCML.
// ============================================================================

typedef __bf16 bf16_t;
typedef bf16_t v16bf __attribute__((ext_vector_type(16)));
typedef float  v8f   __attribute__((ext_vector_type(8)));
typedef unsigned int v4u __attribute__((ext_vector_type(4)));
typedef int v4i __attribute__((ext_vector_type(4)));
typedef int v8i __attribute__((ext_vector_type(8)));

// ---------------- LDS layout (bytes) ----------------
#define OFF_ACT0   0        // bf16 128x128 (also 128x64) : layer input ping
#define OFF_ACT1   32768    // bf16 128x128 : layer pong
#define OFF_Y      65536    // f32 128x128 : y carry (CDE: 128x64)
#define OFF_Z      131072   // f32 128x128 : z carry
#define OFF_F      196608   // bf16 128x128 : drift f / CDE field g
#define OFF_E0     229376   // bf16 128x128 : einsum(g,dw_next)   [CDE: CF1 weights]
#define OFF_E1     262144   // bf16 128x128 : einsum(g,dw_cur)    [CDE: CF2 weights]
#define OFF_DW0    294912   // f32 128x16 : dw_j                  [CDE: CF0 weights]
#define OFF_DW1    303104   // f32 128x16 : dw_j+1
#define OFF_MISC   311296   // f32 misc
#define SMEM_BYTES 314368

#define MI_C0   0
#define MI_B2   128
#define MI_DRC  256
#define MI_DFC  384
#define MI_TS   512
#define MI_XT   576

// ---------------- bf16 weight arena offsets in d_ws (elements) --------------
#define WO_CI1 0        // 128x128
#define WO_CI2 16384    //  64x128
#define WO_CF0 24576    // 128x64
#define WO_CF1 32768    // 128x128
#define WO_CF2 49152    // 128x128
#define WO_D0  65536    // 128x128 (x-columns of 128x130)
#define WO_D1  81920
#define WO_D2  98304
#define WO_G0  114688
#define WO_G1  131072
#define WO_G2  147456   // 2048x128
#define WS_ELEMS 409600

struct KParams {
  const float *ts, *paths, *dW;
  const float *init_W, *init_b;
  const float *dr_W0, *dr_b0, *dr_b1, *dr_b2;
  const float *df_W0, *df_b0, *df_b1, *df_b2;
  const float *ro_W, *ro_b;
  const float *ci_W0, *ci_b0, *ci_b1, *ci_b2;
  const float *cf_W0, *cf_b0, *cf_b1, *cf_b2;
  const float *cro_W, *cro_b;
  float *out;
};

// ---------------- fast math: v_rcp_f32 / v_tanh_f32 ----------------
__device__ __forceinline__ float fast_rcp(float x) {
#if __has_builtin(__builtin_amdgcn_rcpf)
  return __builtin_amdgcn_rcpf(x);
#else
  return 1.0f / x;
#endif
}
__device__ __forceinline__ float lipswish_f(float x) {
  return 0.909f * x * fast_rcp(1.0f + __expf(-x));
}
__device__ __forceinline__ float fast_tanh(float x) {
#if __has_builtin(__builtin_amdgcn_tanhf)
  return __builtin_amdgcn_tanhf(x);      // v_tanh_f32
#else
  const float t = __expf(-2.0f * fabsf(x));
  const float r = (1.0f - t) * fast_rcp(1.0f + t);
  return copysignf(r, x);
#endif
}

// ---------------- Tensor Data Mover 1-D DMA (global -> LDS) ----------------
#if __has_builtin(__builtin_amdgcn_tensor_load_to_lds) && \
    __has_builtin(__builtin_amdgcn_s_wait_tensorcnt)
#define HAVE_TDM 1
#else
#define HAVE_TDM 0
#endif

#if HAVE_TDM
// D# per CDNA5 ISA: group0 = {count|flags, lds_addr,
// global_addr[31:0], global_addr[56:32]|type=2}; group1 carries data_size,
// tensor_dim0/1, tile_dim0/1.  1-D tile => straight DMA memcpy.
__device__ __forceinline__ void tdm_copy_1d(unsigned ldsByteOff, const void* gsrc,
                                            unsigned nElem, int dataSizeCode) {
  const unsigned long long ga = (unsigned long long)(uintptr_t)gsrc;
  v4u g0;
  g0.x = 1u;                                    // count=1 (valid), user mode
  g0.y = ldsByteOff;                            // lds_addr (bytes)
  g0.z = (unsigned)(ga & 0xffffffffu);          // global_addr[31:0]
  g0.w = (unsigned)((ga >> 32) & 0x01ffffffu) | (2u << 30);  // [56:32] | type=2
  v8i g1;
  g1[0] = dataSizeCode << 16;                   // wg_mask=0 (not in cluster), data_size
  g1[1] = (int)((nElem & 0xffffu) << 16);       // tensor_dim0[15:0]
  g1[2] = (int)(((nElem >> 16) & 0xffffu) | (1u << 16)); // tensor_dim0[31:16], tensor_dim1=1
  g1[3] = (int)((nElem & 0xffffu) << 16);       // tile_dim0
  g1[4] = 1;                                    // tile_dim1=1
  g1[5] = 0; g1[6] = 0; g1[7] = 0;              // strides unused for 1-D
  const v4i z = {0, 0, 0, 0};
#if __clang_major__ >= 23
  const v8i z8 = {0, 0, 0, 0, 0, 0, 0, 0};
  __builtin_amdgcn_tensor_load_to_lds(g0, g1, z, z, z8, 0);
#else
  __builtin_amdgcn_tensor_load_to_lds(g0, g1, z, z, 0);
#endif
}
#endif

__device__ __forceinline__ v8f wmma_bf16(v16bf a, v16bf b, v8f c) {
  return __builtin_amdgcn_wmma_f32_16x16x32_bf16(false, a, false, b,
                                                 (short)0, c, false, false);
}

// A fragment: 16x32 bf16 per ISA A-table (lane half -> K+0/K+8, VGPR 4-7 -> K+16).
__device__ __forceinline__ v16bf frag_a(const bf16_t* A, int lda, int m0, int k0, int lane) {
  union { unsigned int u[8]; v16bf v; } r;
  const bf16_t* row = A + (m0 + (lane & 15)) * lda;
  const int kb = k0 + ((lane >> 4) << 3);
#pragma unroll
  for (int i = 0; i < 8; ++i) {
    const int kk = kb + ((i < 4) ? (i << 1) : (16 + ((i - 4) << 1)));
    r.u[i] = *(const unsigned int*)(row + kk);
  }
  return r.v;
}

// B fragment: 32x16 bf16; same K order as A (lane half -> K+0/K+8, VGPR 4-7 -> K+16); row-major
// (out,in) weight array IS B column-major -> direct dword loads.
__device__ __forceinline__ v16bf frag_b(const bf16_t* W, int ldk, int n0, int k0, int lane) {
  union { unsigned int u[8]; v16bf v; } r;
  const bf16_t* row = W + (long)(n0 + (lane & 15)) * ldk;
  const int kb = k0 + ((lane >> 4) << 3);
#pragma unroll
  for (int i = 0; i < 8; ++i) r.u[i] = *(const unsigned int*)(row + kb + ((i < 4) ? (i << 1) : (16 + ((i - 4) << 1))));
  return r.v;
}

// Generic 128xN WMMA linear layer. act: 0=none, 1=lipswish, 2=tanh.
__device__ void linear_layer(const bf16_t* __restrict__ A, int lda, int K,
                             const bf16_t* __restrict__ W,
                             const float* __restrict__ bias,
                             const float* __restrict__ ccol,
                             const float* __restrict__ c0,
                             int N, int act,
                             bf16_t* __restrict__ outB, int ldo,
                             float* __restrict__ outF, float* __restrict__ outF2, int ldf,
                             int wave, int lane) {
  const int ntiles = 8 * (N >> 4);
  for (int t = wave; t < ntiles; t += 8) {
    const int tm = t & 7, tn = t >> 3;
    v8f acc = {};
    for (int k0 = 0; k0 < K; k0 += 32) {
      const v16bf a = frag_a(A, lda, tm << 4, k0, lane);
      const v16bf b = frag_b(W, K, tn << 4, k0, lane);
      acc = wmma_bf16(a, b, acc);
    }
    const int nl = lane & 15;
    const int n  = (tn << 4) + nl;
    const float bn = bias ? bias[n] : 0.0f;
    const float wc = ccol ? ccol[n] : 0.0f;
    const int mb = (tm << 4) + ((lane >> 4) << 3);
#pragma unroll
    for (int r = 0; r < 8; ++r) {
      const int m = mb + r;
      float v = acc[r] + bn;
      if (ccol) v += wc * c0[m];
      if (act == 1) v = lipswish_f(v);
      else if (act == 2) v = fast_tanh(v);
      if (outB)  outB[m * ldo + n] = (bf16_t)v;
      if (outF)  outF[m * ldf + n] = v;
      if (outF2) outF2[m * ldf + n] = v;
    }
  }
}

// Diffusion output layer 128x128x2048 with fused noise contraction for
// dw_j (->eCur) and dw_{j+1} (->eNxt); sqrt(dt) folded here so dW is DMA'd raw.
__device__ void diff_big_layer(const bf16_t* __restrict__ A,
                               const bf16_t* __restrict__ W,
                               const float* __restrict__ b2,
                               const float* __restrict__ dwc,
                               const float* __restrict__ dwn,
                               float sdt,
                               bf16_t* __restrict__ eCur, bf16_t* __restrict__ eNxt,
                               int wave, int lane) {
  const int nl = lane & 15;
  for (int t = wave; t < 1024; t += 8) {
    const int tm = t & 7, h = t >> 3;
    if (t + 8 < 1024)  // global_prefetch next tile's B rows
      __builtin_prefetch(W + (long)(((t + 8) >> 3) * 16 + nl) * 128, 0, 0);
    v8f acc = {};
    for (int k0 = 0; k0 < 128; k0 += 32) {
      const v16bf a = frag_a(A, 128, tm << 4, k0, lane);
      const v16bf b = frag_b(W, 128, h << 4, k0, lane);
      acc = wmma_bf16(a, b, acc);
    }
    const float bn = b2[(h << 4) + nl];
    const int mb = (tm << 4) + ((lane >> 4) << 3);
    float pc[8], pn[8];
#pragma unroll
    for (int r = 0; r < 8; ++r) {
      const int m = mb + r;
      const float v = fast_tanh(acc[r] + bn);
      pc[r] = v * dwc[m * 16 + nl];
      pn[r] = v * dwn[m * 16 + nl];
    }
#pragma unroll
    for (int r = 0; r < 8; ++r) {
#pragma unroll
      for (int s = 1; s < 16; s <<= 1) {
        pc[r] += __shfl_xor(pc[r], s, 32);
        pn[r] += __shfl_xor(pn[r], s, 32);
      }
    }
    if (nl == 0) {
#pragma unroll
      for (int r = 0; r < 8; ++r) {
        const int m = mb + r;
        eCur[m * 128 + h] = (bf16_t)(pc[r] * sdt);
        eNxt[m * 128 + h] = (bf16_t)(pn[r] * sdt);
      }
    }
  }
}

// First two (128-wide, lipswish) layers of a 3-layer MLP: ACT0 -> ACT1 -> ACT0.
__device__ void mlp_front(char* smem, const bf16_t* w0, const bf16_t* w1,
                          const float* bias0, const float* b1,
                          const float* ccol, const float* c0, int K0,
                          int wave, int lane) {
  bf16_t* A0 = (bf16_t*)(smem + OFF_ACT0);
  bf16_t* A1 = (bf16_t*)(smem + OFF_ACT1);
  linear_layer(A0, K0, K0, w0, bias0, ccol, c0, 128, 1, A1, 128, nullptr, nullptr, 0, wave, lane);
  __syncthreads();
  linear_layer(A1, 128, 128, w1, b1, nullptr, nullptr, 128, 1, A0, 128, nullptr, nullptr, 0, wave, lane);
  __syncthreads();
}

// ---------------------------------------------------------------------------
// weight conversion fp32 -> padded bf16 arena
// ---------------------------------------------------------------------------
struct ConvEnt { const float* src; int ld, off, K, N; long dst; };
struct ConvTab { ConvEnt e[11]; };

__global__ void convert_weights_kernel(ConvTab T, bf16_t* __restrict__ ws) {
  long idx = (long)blockIdx.x * blockDim.x + threadIdx.x;
  for (int i = 0; i < 11; ++i) {
    const long n = (long)T.e[i].K * T.e[i].N;
    if (idx < n) {
      const int k = (int)(idx % T.e[i].K);
      const int o = (int)(idx / T.e[i].K);
      ws[T.e[i].dst + idx] = (bf16_t)T.e[i].src[(long)o * T.e[i].ld + T.e[i].off + k];
      return;
    }
    idx -= n;
  }
}

// ---------------------------------------------------------------------------
// persistent kernel: full CDE + SDE scans, one workgroup, 8 wave32s
// ---------------------------------------------------------------------------
__global__ __launch_bounds__(256)
void pathcde_persistent_kernel(KParams P, const bf16_t* __restrict__ ws) {
  extern __shared__ char smem[];
  bf16_t* A0  = (bf16_t*)(smem + OFF_ACT0);
  bf16_t* A1  = (bf16_t*)(smem + OFF_ACT1);
  float*  Yb  = (float*)(smem + OFF_Y);
  float*  Zb  = (float*)(smem + OFF_Z);
  bf16_t* Fb  = (bf16_t*)(smem + OFF_F);
  bf16_t* E0  = (bf16_t*)(smem + OFF_E0);
  bf16_t* E1  = (bf16_t*)(smem + OFF_E1);
  float*  DW0 = (float*)(smem + OFF_DW0);
  float*  DW1 = (float*)(smem + OFF_DW1);
  float*  MISC = (float*)(smem + OFF_MISC);
  float* C0v = MISC + MI_C0;
  float* B2  = MISC + MI_B2;
  float* DRC = MISC + MI_DRC;
  float* DFC = MISC + MI_DFC;
  float* TSL = MISC + MI_TS;
  float* XTv = MISC + MI_XT;

  const int tid = threadIdx.x;
  const int wave = tid >> 5, lane = tid & 31;

  // CDE-phase LDS weight copies overlay the (idle) E/DW regions.
  bf16_t* CF0L = (bf16_t*)(smem + OFF_DW0);    // 128x64  (16KB)
  bf16_t* CF1L = (bf16_t*)(smem + OFF_E0);     // 128x128 (32KB)
  bf16_t* CF2L = (bf16_t*)(smem + OFF_E1);     // 128x128 (32KB)

#if HAVE_TDM
  if (wave == 0) {                             // TDM DMA: global(ws) -> LDS
    tdm_copy_1d(OFF_DW0, ws + WO_CF0, 8192, 1);
    tdm_copy_1d(OFF_E0,  ws + WO_CF1, 16384, 1);
    tdm_copy_1d(OFF_E1,  ws + WO_CF2, 16384, 1);
    __builtin_amdgcn_s_wait_tensorcnt(0);
  }
#else
  for (int i = tid; i < 1024; i += 256)
    ((uint4*)CF0L)[i] = ((const uint4*)(ws + WO_CF0))[i];
  for (int i = tid; i < 2048; i += 256) {
    ((uint4*)CF1L)[i] = ((const uint4*)(ws + WO_CF1))[i];
    ((uint4*)CF2L)[i] = ((const uint4*)(ws + WO_CF2))[i];
  }
#endif

  if (tid < 64) TSL[tid] = P.ts[tid];
  if (tid < 128) {
    XTv[tid] = P.paths[((tid >> 3) * 64 + 63) * 2 + 1];  // xT = paths_r[:,-1,1]
    DRC[tid] = P.dr_W0[tid * 130 + 129];                 // c0 columns of layer-0 W
    DFC[tid] = P.df_W0[tid * 130 + 129];
  }

  // ======================= CDE init MLP  (h0) ==============================
  for (int i = tid; i < 128 * 128; i += 256) {
    const int m = i >> 7, o = i & 127, b = m >> 3;
    const float p0 = P.paths[b * 128 + 0];
    const float p1 = P.paths[b * 128 + 1];
    const float v = P.ci_W0[o * 2 + 0] * p0 + P.ci_W0[o * 2 + 1] * p1 + P.ci_b0[o];
    A0[m * 128 + o] = (bf16_t)lipswish_f(v);
  }
  __syncthreads();
  linear_layer(A0, 128, 128, ws + WO_CI1, P.ci_b1, nullptr, nullptr, 128, 1,
               A1, 128, nullptr, nullptr, 0, wave, lane);
  __syncthreads();
  linear_layer(A1, 128, 128, ws + WO_CI2, P.ci_b2, nullptr, nullptr, 64, 0,
               A0, 64, Yb, Zb, 64, wave, lane);
  __syncthreads();

  // g0 = field(t=0, h0) -> Fb
  if (tid < 128) B2[tid] = P.cf_b0[tid];
  __syncthreads();
  mlp_front(smem, CF0L, CF1L, B2, P.cf_b1, nullptr, nullptr, 64, wave, lane);
  linear_layer(A0, 128, 128, CF2L, P.cf_b2, nullptr, nullptr, 128, 2,
               Fb, 128, nullptr, nullptr, 0, wave, lane);
  __syncthreads();

  // ============================ CDE scan ===================================
  for (int j = 0; j < 63; ++j) {
    const float t1 = (float)(j + 1);
    for (int i = tid; i < 128 * 64; i += 256) {
      const int m = i >> 6, h = i & 63, b = m >> 3;
      const float dx0 = P.paths[(b * 64 + j + 1) * 2 + 0] - P.paths[(b * 64 + j) * 2 + 0];
      const float dx1 = P.paths[(b * 64 + j + 1) * 2 + 1] - P.paths[(b * 64 + j) * 2 + 1];
      const float e = (float)Fb[m * 128 + 2 * h] * dx0 + (float)Fb[m * 128 + 2 * h + 1] * dx1;
      const float y = Yb[i], z = Zb[i];
      const float z1 = 2.0f * y - z + e;
      Zb[i] = z1;
      Yb[i] = y + 0.5f * e;
      A0[i] = (bf16_t)z1;   // field input, ld 64
    }
    if (tid < 128) B2[tid] = P.cf_b0[tid] + t1 * P.cf_W0[tid * 65]; // fold t into bias
    __syncthreads();
    mlp_front(smem, CF0L, CF1L, B2, P.cf_b1, nullptr, nullptr, 64, wave, lane);
    linear_layer(A0, 128, 128, CF2L, P.cf_b2, nullptr, nullptr, 128, 2,
                 Fb, 128, nullptr, nullptr, 0, wave, lane);
    __syncthreads();
    for (int i = tid; i < 128 * 64; i += 256) {
      const int m = i >> 6, h = i & 63, b = m >> 3;
      const float dx0 = P.paths[(b * 64 + j + 1) * 2 + 0] - P.paths[(b * 64 + j) * 2 + 0];
      const float dx1 = P.paths[(b * 64 + j + 1) * 2 + 1] - P.paths[(b * 64 + j) * 2 + 1];
      Yb[i] += 0.5f * ((float)Fb[m * 128 + 2 * h] * dx0 + (float)Fb[m * 128 + 2 * h + 1] * dx1);
    }
    __syncthreads();
  }

  // c0 = hT @ cde_ro_W.T + b
  if (tid < 128) {
    float s = P.cro_b[0];
    for (int k = 0; k < 64; ++k) s += Yb[tid * 64 + k] * P.cro_W[k];
    C0v[tid] = s;
  }
  __syncthreads();

  // x0 = [xT, c0] @ init_W.T + init_b  -> Y, Z
  for (int i = tid; i < 128 * 128; i += 256) {
    const int m = i >> 7, o = i & 127;
    const float v = P.init_W[o * 2 + 0] * XTv[m] + P.init_W[o * 2 + 1] * C0v[m] + P.init_b[o];
    Yb[i] = v; Zb[i] = v;
  }
  __syncthreads();

  // output: ts channel for all l, ys for l=0
  if (tid < 128) {
    float s = P.ro_b[0] + C0v[tid] * P.ro_W[128];
    for (int k = 0; k < 128; ++k) s += Yb[tid * 128 + k] * P.ro_W[k];
    for (int l = 0; l < 64; ++l) P.out[(tid * 64 + l) * 2 + 0] = TSL[l];
    P.out[(tid * 64 + 0) * 2 + 1] = s;
  }
  __syncthreads();

  const float dt = TSL[1] - TSL[0];
  const float sdt = sqrtf(dt);

  // ================= SDE init: f0 -> Fb, e_old = einsum(g0,dw0) -> E0 ======
  for (int i = tid; i < 128 * 128; i += 256) A0[i] = (bf16_t)Zb[i];
  if (tid < 128) B2[tid] = P.dr_b0[tid] + TSL[0] * P.dr_W0[tid * 130];
  __syncthreads();
  mlp_front(smem, ws + WO_D0, ws + WO_D1, B2, P.dr_b1, DRC, C0v, 128, wave, lane);
  linear_layer(A0, 128, 128, ws + WO_D2, P.dr_b2, nullptr, nullptr, 128, 2,
               Fb, 128, nullptr, nullptr, 0, wave, lane);
  __syncthreads();

  for (int i = tid; i < 128 * 128; i += 256) A0[i] = (bf16_t)Zb[i];
  if (tid < 128) B2[tid] = P.df_b0[tid] + TSL[0] * P.df_W0[tid * 130];
#if HAVE_TDM
  if (wave == 0) {                             // raw dW_0 into both slots
    tdm_copy_1d(OFF_DW0, P.dW, 2048, 2);
    tdm_copy_1d(OFF_DW1, P.dW, 2048, 2);
  }
#else
  for (int i = tid; i < 2048; i += 256) {
    const float w = P.dW[i];
    DW0[i] = w; DW1[i] = w;
  }
#endif
  __syncthreads();
  mlp_front(smem, ws + WO_G0, ws + WO_G1, B2, P.df_b1, DFC, C0v, 128, wave, lane);
#if HAVE_TDM
  if (wave == 0) __builtin_amdgcn_s_wait_tensorcnt(0);
  __syncthreads();
#endif
  diff_big_layer(A0, ws + WO_G2, P.df_b2, DW0, DW1, sdt, E1, E0, wave, lane);
  __syncthreads();

  // ============================ SDE scan ===================================
  for (int j = 0; j < 63; ++j) {
    const float t1 = TSL[j + 1];
    const int jn = (j + 1 < 63) ? (j + 1) : 62;
#if HAVE_TDM
    if (wave == 0) {                           // DMA this step's dW pair early
      tdm_copy_1d(OFF_DW0, P.dW + (long)j * 2048, 2048, 2);
      tdm_copy_1d(OFF_DW1, P.dW + (long)jn * 2048, 2048, 2);
    }
#endif
    // z1 = 2y - z + f*dt + e_old ; ypre = y + 0.5*f*dt + 0.5*e_old
    for (int i = tid; i < 128 * 128; i += 256) {
      const float y = Yb[i], z = Zb[i];
      const float f = (float)Fb[i];
      const float e = (float)E0[i];
      const float z1 = 2.0f * y - z + f * dt + e;
      Zb[i] = z1;
      Yb[i] = y + 0.5f * f * dt + 0.5f * e;
      A0[i] = (bf16_t)z1;   // drift input copy
    }
    if (tid < 128) B2[tid] = P.dr_b0[tid] + t1 * P.dr_W0[tid * 130];
    __syncthreads();
    mlp_front(smem, ws + WO_D0, ws + WO_D1, B2, P.dr_b1, DRC, C0v, 128, wave, lane);
    linear_layer(A0, 128, 128, ws + WO_D2, P.dr_b2, nullptr, nullptr, 128, 2,
                 Fb, 128, nullptr, nullptr, 0, wave, lane);  // f1
    __syncthreads();

    for (int i = tid; i < 128 * 128; i += 256) A0[i] = (bf16_t)Zb[i];
    if (tid < 128) B2[tid] = P.df_b0[tid] + t1 * P.df_W0[tid * 130];
#if !HAVE_TDM
    for (int i = tid; i < 2048; i += 256) {
      DW0[i] = P.dW[(long)j * 2048 + i];
      DW1[i] = P.dW[(long)jn * 2048 + i];
    }
#endif
    __syncthreads();
    mlp_front(smem, ws + WO_G0, ws + WO_G1, B2, P.df_b1, DFC, C0v, 128, wave, lane);
#if HAVE_TDM
    if (wave == 0) __builtin_amdgcn_s_wait_tensorcnt(0);
    __syncthreads();
#endif
    diff_big_layer(A0, ws + WO_G2, P.df_b2, DW0, DW1, sdt, E1, E0, wave, lane);
    __syncthreads();

    // y1 = ypre + 0.5*f1*dt + 0.5*einsum(g1,dw_j)
    for (int i = tid; i < 128 * 128; i += 256)
      Yb[i] += 0.5f * (float)Fb[i] * dt + 0.5f * (float)E1[i];
    __syncthreads();
    if (tid < 128) {
      float s = P.ro_b[0] + C0v[tid] * P.ro_W[128];
      for (int k = 0; k < 128; ++k) s += Yb[tid * 128 + k] * P.ro_W[k];
      P.out[(tid * 64 + (j + 1)) * 2 + 1] = s;
    }
    __syncthreads();
  }
}

// ---------------------------------------------------------------------------
// host launcher
// ---------------------------------------------------------------------------
extern "C" void kernel_launch(void* const* d_in, const int* in_sizes, int n_in,
                              void* d_out, int out_size, void* d_ws, size_t ws_size,
                              hipStream_t stream) {
  (void)in_sizes; (void)n_in; (void)out_size; (void)ws_size;
  bf16_t* ws = (bf16_t*)d_ws;

  ConvTab T;
  auto set = [&](int i, const void* src, int ld, int off, int K, int N, long dst) {
    T.e[i].src = (const float*)src; T.e[i].ld = ld; T.e[i].off = off;
    T.e[i].K = K; T.e[i].N = N; T.e[i].dst = dst;
  };
  set(0,  d_in[21], 128, 0, 128, 128,  WO_CI1);  // cde_init W1
  set(1,  d_in[23], 128, 0, 128, 64,   WO_CI2);  // cde_init W2
  set(2,  d_in[25], 65,  1, 64,  128,  WO_CF0);  // cde_field W0 (drop t col)
  set(3,  d_in[27], 128, 0, 128, 128,  WO_CF1);
  set(4,  d_in[29], 128, 0, 128, 128,  WO_CF2);
  set(5,  d_in[5],  130, 1, 128, 128,  WO_D0);   // drift W0 (drop t,c0 cols)
  set(6,  d_in[7],  128, 0, 128, 128,  WO_D1);
  set(7,  d_in[9],  128, 0, 128, 128,  WO_D2);
  set(8,  d_in[11], 130, 1, 128, 128,  WO_G0);   // diff W0
  set(9,  d_in[13], 128, 0, 128, 128,  WO_G1);
  set(10, d_in[15], 128, 0, 128, 2048, WO_G2);   // diff W2 (big)
  convert_weights_kernel<<<(WS_ELEMS + 255) / 256, 256, 0, stream>>>(T, ws);

  KParams P;
  P.ts = (const float*)d_in[0];   P.paths = (const float*)d_in[1];
  P.dW = (const float*)d_in[2];
  P.init_W = (const float*)d_in[3]; P.init_b = (const float*)d_in[4];
  P.dr_W0 = (const float*)d_in[5];  P.dr_b0 = (const float*)d_in[6];
  P.dr_b1 = (const float*)d_in[8];  P.dr_b2 = (const float*)d_in[10];
  P.df_W0 = (const float*)d_in[11]; P.df_b0 = (const float*)d_in[12];
  P.df_b1 = (const float*)d_in[14]; P.df_b2 = (const float*)d_in[16];
  P.ro_W = (const float*)d_in[17];  P.ro_b = (const float*)d_in[18];
  P.ci_W0 = (const float*)d_in[19]; P.ci_b0 = (const float*)d_in[20];
  P.ci_b1 = (const float*)d_in[22]; P.ci_b2 = (const float*)d_in[24];
  P.cf_W0 = (const float*)d_in[25]; P.cf_b0 = (const float*)d_in[26];
  P.cf_b1 = (const float*)d_in[28]; P.cf_b2 = (const float*)d_in[30];
  P.cro_W = (const float*)d_in[31]; P.cro_b = (const float*)d_in[32];
  P.out = (float*)d_out;

  pathcde_persistent_kernel<<<1, 256, SMEM_BYTES, stream>>>(P, ws);
}
